// Attention_979252543724
// MI455X (gfx1250) — compile-verified, matrix operand order checked by reading, not run on hardware
//
#include <hip/hip_runtime.h>
#include <hip/hip_bf16.h>

// Problem constants from the reference: B=2, T=2048, D=1024, H=16, DH=64.
#define BB 2
#define TT 2048
#define DD 1024
#define HH 16
#define DH 64

typedef __bf16 bf16;
typedef __attribute__((ext_vector_type(16))) __bf16 v16bf;
typedef __attribute__((ext_vector_type(8)))  __bf16 v8bf;
typedef __attribute__((ext_vector_type(8)))  float  v8f;

// ---------------------------------------------------------------------------
// WMMA helpers (CDNA5 / gfx1250, wave32)
// ---------------------------------------------------------------------------
__device__ __forceinline__ v8f wmma_bf16(v16bf a, v16bf b, v8f c) {
    // D = A(16x32 bf16) * B(32x16 bf16) + C(16x16 f32)
    return __builtin_amdgcn_wmma_f32_16x16x32_bf16(
        /*neg_a=*/false, a, /*neg_b=*/false, b,
        /*c_mod=*/(short)0, c, /*reuse_a=*/false, /*reuse_b=*/false);
}

// Load a 16x32 bf16 fragment from a row-major [rows, ld] bf16 matrix, k-chunk kc.
// Matches the ISA 16-bit A-matrix layout (lanes 0-15: K=0-7 in v0-3, K=16-23 in
// v4-7; lanes 16-31: K=8-15, K=24-31); by symmetry it is also the B-matrix
// layout when the B source is stored [N, K] row-major (we compute A * Bsrc^T).
// Requires base 16B aligned, ld % 8 == 0, kc % 32 == 0.
__device__ __forceinline__ v16bf load_frag(const bf16* __restrict__ base, int ld, int kc) {
    const int lane = threadIdx.x & 31;
    const int row  = lane & 15;
    const int koff = kc + ((lane >> 4) << 3);   // 0 or 8
    const bf16* p = base + (size_t)row * ld + koff;
    v8bf lo = *reinterpret_cast<const v8bf*>(p);        // K = koff + 0..7
    v8bf hi = *reinterpret_cast<const v8bf*>(p + 16);   // K = koff + 16..23
    return __builtin_shufflevector(lo, hi, 0,1,2,3,4,5,6,7,8,9,10,11,12,13,14,15);
}

// ---------------------------------------------------------------------------
// Prep kernels
// ---------------------------------------------------------------------------
__global__ void cvt_f32_to_bf16_kernel(const float* __restrict__ src,
                                       bf16* __restrict__ dst, int n) {
    int i = blockIdx.x * blockDim.x + threadIdx.x;
    if (i < n) dst[i] = (bf16)src[i];
}

// src: f32 [K, N] row-major  ->  dst: bf16 [N, K] row-major
__global__ void transpose_cvt_kernel(const float* __restrict__ src,
                                     bf16* __restrict__ dst, int K, int N) {
    int i = blockIdx.x * blockDim.x + threadIdx.x;
    if (i >= K * N) return;
    int k = i / N, n = i - k * N;
    dst[(size_t)n * K + k] = (bf16)src[i];
}

// ---------------------------------------------------------------------------
// bf16 GEMM: C[M,N] = A[M,K] * Bt[N,K]^T   (fp32 accumulate)
// Block = 256 threads = 8 waves; wave tile 64x64 (16 fp32 accumulators,
// 1.0 b128-load per WMMA); block tile 128x256.
// Store fp32 (Cf) or bf16 (Cb), whichever pointer is non-null.
// ---------------------------------------------------------------------------
__global__ __launch_bounds__(256)
void gemm_bf16_kernel(const bf16* __restrict__ A, const bf16* __restrict__ Bt,
                      float* __restrict__ Cf, bf16* __restrict__ Cb,
                      int M, int N, int Kd) {
    const int wave = threadIdx.x >> 5;
    const int lane = threadIdx.x & 31;
    const int m0 = blockIdx.x * 128 + (wave & 1) * 64;
    const int n0 = blockIdx.y * 256 + (wave >> 1) * 64;

    v8f acc[4][4];
#pragma unroll
    for (int i = 0; i < 4; ++i)
#pragma unroll
        for (int j = 0; j < 4; ++j)
#pragma unroll
            for (int v = 0; v < 8; ++v) acc[i][j][v] = 0.0f;

    const bf16* Arows[4];
#pragma unroll
    for (int i = 0; i < 4; ++i) Arows[i] = A + (size_t)(m0 + i * 16) * Kd;

    for (int kc = 0; kc < Kd; kc += 32) {
        v16bf a[4];
#pragma unroll
        for (int i = 0; i < 4; ++i) a[i] = load_frag(Arows[i], Kd, kc);
#pragma unroll
        for (int j = 0; j < 4; ++j) {
            v16bf bf = load_frag(Bt + (size_t)(n0 + j * 16) * Kd, Kd, kc);
#pragma unroll
            for (int i = 0; i < 4; ++i)
                acc[i][j] = wmma_bf16(a[i], bf, acc[i][j]);
        }
    }

    // C layout: VGPR v holds row (v + 8*(lane>>4)), column (lane & 15).
    const int cg = lane >> 4, cn = lane & 15;
#pragma unroll
    for (int i = 0; i < 4; ++i)
#pragma unroll
        for (int j = 0; j < 4; ++j)
#pragma unroll
            for (int v = 0; v < 8; ++v) {
                int row = m0 + i * 16 + v + 8 * cg;
                int col = n0 + j * 16 + cn;
                size_t idx = (size_t)row * N + col;
                if (Cf) Cf[idx] = acc[i][j][v];
                else    Cb[idx] = (bf16)acc[i][j][v];
            }
}

// ---------------------------------------------------------------------------
// RoPE + head split. qkvb: bf16 [B*T, 3*D]. One thread per (b,t,h,i), i<32.
// Writes Q,K (rotary applied): bf16 [B,H,T,DH]; V transposed: bf16 [B,H,DH,T].
// ---------------------------------------------------------------------------
__global__ void rope_split_kernel(const bf16* __restrict__ qkvb,
                                  const float* __restrict__ cosT,
                                  const float* __restrict__ sinT,
                                  bf16* __restrict__ Q, bf16* __restrict__ K,
                                  bf16* __restrict__ Vt) {
    int idx = blockIdx.x * blockDim.x + threadIdx.x;
    if (idx >= BB * TT * HH * 32) return;
    int i = idx & 31;
    int h = (idx >> 5) & (HH - 1);
    int t = (idx >> 9) & (TT - 1);
    int b = idx >> 20;

    const bf16* row = qkvb + (size_t)(b * TT + t) * (3 * DD);
    float c = cosT[t * 32 + i];
    float s = sinT[t * 32 + i];
    int qc = h * DH;

    float q1 = (float)row[qc + i],          q2 = (float)row[qc + 32 + i];
    float k1 = (float)row[DD + qc + i],     k2 = (float)row[DD + qc + 32 + i];
    float v1 = (float)row[2 * DD + qc + i], v2 = (float)row[2 * DD + qc + 32 + i];

    size_t o = ((size_t)(b * HH + h) * TT + t) * DH;
    Q[o + i]      = (bf16)(q1 * c - q2 * s);
    Q[o + 32 + i] = (bf16)(q2 * c + q1 * s);
    K[o + i]      = (bf16)(k1 * c - k2 * s);
    K[o + 32 + i] = (bf16)(k2 * c + k1 * s);

    size_t vo = (size_t)(b * HH + h) * DH * TT + t;   // Vt[(bh*DH + dh)*T + t]
    Vt[vo + (size_t)i * TT]        = (bf16)v1;
    Vt[vo + (size_t)(i + 32) * TT] = (bf16)v2;
}

// ---------------------------------------------------------------------------
// Causal flash attention. One wave = 16 query rows x full DH=64 output.
// Block = 128 threads (4 waves, 64 query rows). Grid = (T/64, H, B).
// 64-key chunks: 8 WMMAs for S(16x64), 8 WMMAs for P@V per softmax pass.
// Online softmax in fp32 with cross-lane row reductions (16-lane groups).
// ---------------------------------------------------------------------------
__global__ __launch_bounds__(128)
void flash_attn_kernel(const bf16* __restrict__ Q, const bf16* __restrict__ K,
                       const bf16* __restrict__ Vt, bf16* __restrict__ AO) {
    __shared__ bf16 pbuf[4][16 * 64];   // per-wave 16x64 P staging (C->A layout)

    const int wave = threadIdx.x >> 5;
    const int lane = threadIdx.x & 31;
    const int b = blockIdx.z, h = blockIdx.y;
    const int qbase = (blockIdx.x * 4 + wave) * 16;
    const int cg = lane >> 4, cn = lane & 15;

    const size_t hoff = (size_t)(b * HH + h) * TT * DH;   // == (bh)*DH*TT for Vt too
    const bf16* Qh  = Q  + hoff;
    const bf16* Kh  = K  + hoff;
    const bf16* Vth = Vt + hoff;

    v16bf qa0 = load_frag(Qh + (size_t)qbase * DH, DH, 0);
    v16bf qa1 = load_frag(Qh + (size_t)qbase * DH, DH, 32);

    float rmax[8], rsum[8];
    v8f o[4];
#pragma unroll
    for (int v = 0; v < 8; ++v) { rmax[v] = -3.0e38f; rsum[v] = 0.0f; }
#pragma unroll
    for (int t = 0; t < 4; ++t)
#pragma unroll
        for (int v = 0; v < 8; ++v) o[t][v] = 0.0f;

    const float scale = 0.125f;   // 1/sqrt(DH)

    for (int k0 = 0; k0 < qbase + 16; k0 += 64) {
        // ---- S(16x64) = Q(16x64) @ K^T : 4 n-tiles x 2 k-steps ----
        v8f s[4];
#pragma unroll
        for (int nt = 0; nt < 4; ++nt) {
#pragma unroll
            for (int v = 0; v < 8; ++v) s[nt][v] = 0.0f;
            v16bf kb0 = load_frag(Kh + (size_t)(k0 + nt * 16) * DH, DH, 0);
            v16bf kb1 = load_frag(Kh + (size_t)(k0 + nt * 16) * DH, DH, 32);
            s[nt] = wmma_bf16(qa0, kb0, s[nt]);
            s[nt] = wmma_bf16(qa1, kb1, s[nt]);
        }
        // ---- scale + causal mask ----
#pragma unroll
        for (int v = 0; v < 8; ++v) {
            int qrow = qbase + v + 8 * cg;
#pragma unroll
            for (int nt = 0; nt < 4; ++nt) {
                int key = k0 + nt * 16 + cn;
                float sv = s[nt][v] * scale;
                s[nt][v] = (key <= qrow) ? sv : -3.0e38f;
            }
        }
        // ---- online softmax (row reductions across 16-lane groups) ----
#pragma unroll
        for (int v = 0; v < 8; ++v) {
            float m = fmaxf(fmaxf(s[0][v], s[1][v]), fmaxf(s[2][v], s[3][v]));
#pragma unroll
            for (int d = 1; d < 16; d <<= 1) m = fmaxf(m, __shfl_xor(m, d, 32));
            m = fmaxf(m, rmax[v]);
            float corr = __expf(rmax[v] - m);
            rmax[v] = m;
            float p0 = __expf(s[0][v] - m);
            float p1 = __expf(s[1][v] - m);
            float p2 = __expf(s[2][v] - m);
            float p3 = __expf(s[3][v] - m);
            float ps = (p0 + p1) + (p2 + p3);
#pragma unroll
            for (int d = 1; d < 16; d <<= 1) ps += __shfl_xor(ps, d, 32);
            rsum[v] = rsum[v] * corr + ps;
#pragma unroll
            for (int t = 0; t < 4; ++t) o[t][v] *= corr;
            int r = v + 8 * cg;
            pbuf[wave][r * 64 + cn]      = (bf16)p0;
            pbuf[wave][r * 64 + 16 + cn] = (bf16)p1;
            pbuf[wave][r * 64 + 32 + cn] = (bf16)p2;
            pbuf[wave][r * 64 + 48 + cn] = (bf16)p3;
        }
        // Wave-private LDS region; DS ops are in-order per wave. Prevent the
        // compiler from hoisting the loads above the stores:
        asm volatile("" ::: "memory");
        // ---- O(16x64) += P(16x64) @ V(64x64)  (B-src = Vt rows [dh, key]) ----
        v16bf pa0 = load_frag(&pbuf[wave][0], 64, 0);
        v16bf pa1 = load_frag(&pbuf[wave][0], 64, 32);
#pragma unroll
        for (int t = 0; t < 4; ++t) {
            const bf16* vrow = Vth + (size_t)(t * 16) * TT + k0;
            v16bf vb0 = load_frag(vrow, TT, 0);
            v16bf vb1 = load_frag(vrow, TT, 32);
            o[t] = wmma_bf16(pa0, vb0, o[t]);
            o[t] = wmma_bf16(pa1, vb1, o[t]);
        }
    }

    // ---- epilogue: normalize and write AO[B*T, D] (bf16) ----
#pragma unroll
    for (int v = 0; v < 8; ++v) {
        float inv = 1.0f / rsum[v];
        int row = qbase + v + 8 * cg;
#pragma unroll
        for (int t = 0; t < 4; ++t) {
            AO[(size_t)(b * TT + row) * DD + h * DH + t * 16 + cn] =
                (bf16)(o[t][v] * inv);
        }
    }
}

// ---------------------------------------------------------------------------
// Launcher
// ---------------------------------------------------------------------------
extern "C" void kernel_launch(void* const* d_in, const int* in_sizes, int n_in,
                              void* d_out, int out_size, void* d_ws, size_t ws_size,
                              hipStream_t stream) {
    const float* x    = (const float*)d_in[0];   // [B,T,D] f32
    const float* Wqkv = (const float*)d_in[1];   // [D,3D] f32
    const float* Wout = (const float*)d_in[2];   // [D,D] f32
    const float* cosT = (const float*)d_in[3];   // [T,DH/2] f32
    const float* sinT = (const float*)d_in[4];   // [T,DH/2] f32
    // d_in[5] = causal mask: handled analytically.

    float* out = (float*)d_out;                  // [B,T,D] f32

    const int M  = BB * TT;       // 4096
    const int N3 = 3 * DD;        // 3072

    char* ws = (char*)d_ws;
    size_t off = 0;
    bf16* xb    = (bf16*)(ws + off); off += (size_t)M * DD * 2;            // 8 MB
    bf16* WqkvT = (bf16*)(ws + off); off += (size_t)N3 * DD * 2;           // 6 MB
    bf16* WoutT = (bf16*)(ws + off); off += (size_t)DD * DD * 2;           // 2 MB
    bf16* qkvb  = (bf16*)(ws + off); off += (size_t)M * N3 * 2;            // 24 MB
    bf16* Qb    = (bf16*)(ws + off); off += (size_t)BB * HH * TT * DH * 2; // 8 MB
    bf16* Kb    = (bf16*)(ws + off); off += (size_t)BB * HH * TT * DH * 2; // 8 MB
    bf16* Vtb   = (bf16*)(ws + off); off += (size_t)BB * HH * TT * DH * 2; // 8 MB
    bf16* AO    = (bf16*)(ws + off); off += (size_t)M * DD * 2;            // 8 MB

    // 1) convert x -> bf16; transpose+convert weights to [N,K] bf16
    {
        int n = M * DD;
        cvt_f32_to_bf16_kernel<<<(n + 255) / 256, 256, 0, stream>>>(x, xb, n);
        transpose_cvt_kernel<<<(DD * N3 + 255) / 256, 256, 0, stream>>>(Wqkv, WqkvT, DD, N3);
        transpose_cvt_kernel<<<(DD * DD + 255) / 256, 256, 0, stream>>>(Wout, WoutT, DD, DD);
    }
    // 2) qkv = x @ Wqkv   (bf16 out)
    gemm_bf16_kernel<<<dim3(M / 128, N3 / 256), 256, 0, stream>>>(
        xb, WqkvT, nullptr, qkvb, M, N3, DD);
    // 3) RoPE + head split (+ V transpose)
    {
        int n = BB * TT * HH * 32;
        rope_split_kernel<<<(n + 255) / 256, 256, 0, stream>>>(
            qkvb, cosT, sinT, Qb, Kb, Vtb);
    }
    // 4) causal flash attention
    flash_attn_kernel<<<dim3(TT / 64, HH, BB), 128, 0, stream>>>(Qb, Kb, Vtb, AO);
    // 5) out = AO @ Wout   (f32 out)
    gemm_bf16_kernel<<<dim3(M / 128, DD / 256), 256, 0, stream>>>(
        AO, WoutT, out, nullptr, M, DD, DD);
}
